// CausalSelfAttention_13932873908859
// MI455X (gfx1250) — compile-verified, matrix operand order checked by reading, not run on hardware
//
#include <hip/hip_runtime.h>

// ---------------------------------------------------------------------------
// CDNA5 / gfx1250 causal self-attention forward.
// All heavy math on v_wmma_f32_16x16x32_bf16 (8192 MACs/instr, wave32).
// ---------------------------------------------------------------------------

typedef __bf16 bf16;
typedef __attribute__((ext_vector_type(16))) __bf16 v16bf;
typedef __attribute__((ext_vector_type(8)))  __bf16 v8bf;
typedef __attribute__((ext_vector_type(8)))  float  v8f;

#define WMMA_BF16(a, b, c) \
    __builtin_amdgcn_wmma_f32_16x16x32_bf16(false, (a), false, (b), (short)0, (c), false, false)

// ds_swizzle group-of-32 xor mode: imm = (xor<<10) | (or<<5) | and(0x1f)
#define SWZ_XOR(x, imm) \
    __int_as_float(__builtin_amdgcn_ds_swizzle(__float_as_int(x), (imm)))

// Load a 16x32 bf16 A-fragment (or B-fragment from a transposed-stored tile)
// from LDS row-major storage.  Per ISA 7.12.2: lanes 0-15 hold
// K {0..7, 16..23}, lanes 16-31 hold K {8..15, 24..31} (koff=8).
__device__ inline v16bf load_frag(const bf16* row, int koff) {
    union { v16bf f; v8bf h[2]; } u;
    u.h[0] = *(const v8bf*)(row + koff);
    u.h[1] = *(const v8bf*)(row + koff + 16);
    return u.f;
}

// ---------------------------------------------------------------------------
// fp32 -> bf16 converts
// ---------------------------------------------------------------------------
__global__ void __launch_bounds__(256)
f32_to_bf16_kernel(const float* __restrict__ src, bf16* __restrict__ dst, size_t n) {
    for (size_t i = (size_t)blockIdx.x * 256 + threadIdx.x; i < n;
         i += (size_t)gridDim.x * 256)
        dst[i] = (bf16)src[i];
}

// src [K][N] fp32 -> dst [N][K] bf16  (weights pre-transposed so GEMM B-tiles
// stage with the same contiguous pattern as A-tiles)
__global__ void __launch_bounds__(256)
f32_to_bf16_T_kernel(const float* __restrict__ src, bf16* __restrict__ dst,
                     int K, int N) {
    size_t total = (size_t)K * N;
    for (size_t i = (size_t)blockIdx.x * 256 + threadIdx.x; i < total;
         i += (size_t)gridDim.x * 256) {
        int k = (int)(i / N);
        int n = (int)(i % N);
        dst[(size_t)n * K + k] = (bf16)src[i];
    }
}

// ---------------------------------------------------------------------------
// bf16 GEMM:  C[M][ldc] = A[M][K] @ BT[N][K]^T + bias[N]
// Block tile 128x128, K-step 32.  8 waves; each wave owns a 32x64 tile
// (2x4 WMMA accs = 8 v_wmma per K-step against 6 fragment loads).
// ---------------------------------------------------------------------------
template <bool OUT_BF16>
__global__ void __launch_bounds__(256)
gemm_bf16_kernel(const bf16* __restrict__ A, const bf16* __restrict__ BT,
                 const float* __restrict__ bias, void* __restrict__ Cout,
                 int M, int N, int K, int ldc) {
    constexpr int BM = 128, BN = 128, BK = 32;
    constexpr int LDK = BK + 8;                 // 40 halves = 80 B (16B-aligned rows)
    __shared__ bf16 As[BM * LDK];               // 10 KB
    __shared__ bf16 Bs[BN * LDK];               // 10 KB

    const int tid  = threadIdx.x;
    const int lane = tid & 31;
    const int wid  = tid >> 5;
    const int bm   = blockIdx.x * BM;
    const int bn   = blockIdx.y * BN;
    const int waveM = (wid & 3) * 32;           // 4 wave rows
    const int waveN = (wid >> 2) * 64;          // 2 wave cols
    const int rowA  = lane & 15;
    const int koff  = (lane >= 16) ? 8 : 0;

    // each thread stages 2 chunks of A and 2 of B per K-step
    const int sr = tid >> 2;                    // 0..63
    const int sc = (tid & 3) * 8;               // 0,8,16,24

    v8f acc[2][4] = {};

    for (int k0 = 0; k0 < K; k0 += BK) {
        __syncthreads();
#pragma unroll
        for (int t = 0; t < 2; ++t) {           // A tile: 128x32 halves
            int r = sr + t * 64;
            *(v8bf*)&As[r * LDK + sc] =
                *(const v8bf*)&A[(size_t)(bm + r) * K + k0 + sc];
        }
#pragma unroll
        for (int t = 0; t < 2; ++t) {           // B tile: 128x32 halves
            int r = sr + t * 64;
            *(v8bf*)&Bs[r * LDK + sc] =
                *(const v8bf*)&BT[(size_t)(bn + r) * K + k0 + sc];
        }
        if (k0 + BK < K) {                      // keep L2->WGP pipe ahead
            __builtin_prefetch(&A[(size_t)(bm + sr) * K + k0 + BK + sc], 0, 3);
            __builtin_prefetch(&BT[(size_t)(bn + sr) * K + k0 + BK + sc], 0, 3);
        }
        __syncthreads();

        v16bf af[2], bf[4];
#pragma unroll
        for (int i = 0; i < 2; ++i)
            af[i] = load_frag(&As[(waveM + i * 16 + rowA) * LDK], koff);
#pragma unroll
        for (int j = 0; j < 4; ++j)
            bf[j] = load_frag(&Bs[(waveN + j * 16 + rowA) * LDK], koff);
#pragma unroll
        for (int i = 0; i < 2; ++i)
#pragma unroll
            for (int j = 0; j < 4; ++j)
                acc[i][j] = WMMA_BF16(af[i], bf[j], acc[i][j]);
    }

    // C/D layout: lane 0-15 -> N=lane, M=r; lane 16-31 -> N=lane-16, M=r+8
    const int colL = lane & 15;
    const int rsel = (lane >> 4) * 8;
#pragma unroll
    for (int i = 0; i < 2; ++i)
#pragma unroll
        for (int j = 0; j < 4; ++j) {
            int col = bn + waveN + j * 16 + colL;
            float bb = bias ? bias[col] : 0.0f;
#pragma unroll
            for (int r = 0; r < 8; ++r) {
                int row = bm + waveM + i * 16 + rsel + r;
                float v = acc[i][j][r] + bb;
                if (OUT_BF16)
                    ((bf16*)Cout)[(size_t)row * ldc + col] = (bf16)v;
                else
                    ((float*)Cout)[(size_t)row * ldc + col] = v;
            }
        }
}

// ---------------------------------------------------------------------------
// Flash attention (causal).  One block = (b, h, 128-row query tile).
// 8 waves; wave w owns query rows [w*16, w*16+16).  Key/value tiles of 64.
// ---------------------------------------------------------------------------
__global__ void __launch_bounds__(256)
attn_kernel(const bf16* __restrict__ qkv,   // [B*S][3072], cols: q|k|v per head
            bf16* __restrict__ ctx) {       // [B*S][1024]
    constexpr int S = 2048, HD = 64, QT = 128, KT = 64;
    constexpr int LQ = HD + 8, LK = HD + 8, LV = KT + 8, LP = KT + 8;   // 72

    __shared__ bf16 Qs[QT * LQ];        // 18 KB  (pre-scaled by 1/sqrt(hd))
    __shared__ bf16 Ks[KT * LK];        // 9 KB   row-major [key][d]
    __shared__ bf16 VTs[HD * LV];       // 9 KB   transposed [d][key]
    __shared__ bf16 Ps[8 * 16 * LP];    // 18 KB  per-wave P staging

    const int blk = blockIdx.x;
    const int qt = blk & 15;
    const int h  = (blk >> 4) & 15;
    const int b  = blk >> 8;
    const int q0 = qt * QT;

    const size_t rs = 3072;
    const bf16* qbase = qkv + (size_t)(b * S) * rs + h * HD;
    const bf16* kbase = qbase + 1024;
    const bf16* vbase = qbase + 2048;

    const int tid  = threadIdx.x;
    const int lane = tid & 31;
    const int wid  = tid >> 5;
    const int qr0  = wid * 16;
    const int colL = lane & 15;
    const int rsel = (lane >> 4) * 8;
    const int koff = (lane >= 16) ? 8 : 0;

    // Stage Q (scaled by 0.125 = 2^-3, exact in bf16), packed 16B LDS stores
    for (int t = tid; t < QT * 8; t += 256) {
        int r = t >> 3, c = (t & 7) * 8;
        v8bf vq = *(const v8bf*)&qbase[(size_t)(q0 + r) * rs + c];
        v8bf sv;
#pragma unroll
        for (int e = 0; e < 8; ++e) sv[e] = (bf16)((float)vq[e] * 0.125f);
        *(v8bf*)&Qs[r * LQ + c] = sv;
    }

    float m[8], l[8];
#pragma unroll
    for (int r = 0; r < 8; ++r) { m[r] = -3.0e38f; l[r] = 0.0f; }
    v8f o[4] = {};

    const int nkt = q0 / KT + 2;        // key tiles covering causal extent
    for (int kt = 0; kt < nkt; ++kt) {
        __syncthreads();
        const int k0 = kt * KT;
        for (int t = tid; t < KT * 8; t += 256) {       // K tile, row-major
            int r = t >> 3, c = (t & 7) * 8;
            *(v8bf*)&Ks[r * LK + c] =
                *(const v8bf*)&kbase[(size_t)(k0 + r) * rs + c];
        }
        for (int t = tid; t < KT * 8; t += 256) {       // V tile, transposed
            int r = t >> 3, c = (t & 7) * 8;
            v8bf vv = *(const v8bf*)&vbase[(size_t)(k0 + r) * rs + c];
#pragma unroll
            for (int e = 0; e < 8; ++e) VTs[(c + e) * LV + r] = vv[e];
        }
        __syncthreads();

        // scores S = Q @ K^T  (hd=64 -> 2 k-steps per 16x16 tile)
        v8f s[4];
#pragma unroll
        for (int nt = 0; nt < 4; ++nt) {
            v8f z = {};
#pragma unroll
            for (int ks = 0; ks < 2; ++ks) {
                int ko2 = ks * 32 + koff;
                v16bf af = load_frag(&Qs[(qr0 + colL) * LQ], ko2);
                v16bf bf = load_frag(&Ks[(nt * 16 + colL) * LK], ko2);
                z = WMMA_BF16(af, bf, z);
            }
            s[nt] = z;
        }

        // causal mask
#pragma unroll
        for (int nt = 0; nt < 4; ++nt) {
            int kj = k0 + nt * 16 + colL;
#pragma unroll
            for (int r = 0; r < 8; ++r)
                if (kj > q0 + qr0 + rsel + r) s[nt][r] = -3.0e38f;
        }

        // online softmax; row r's 16 values live in lanes 0-15 (or 16-31),
        // so xor-swizzle reductions (1,2,4,8) stay inside each 16-lane half
#pragma unroll
        for (int r = 0; r < 8; ++r) {
            float mx = fmaxf(fmaxf(s[0][r], s[1][r]), fmaxf(s[2][r], s[3][r]));
            mx = fmaxf(mx, SWZ_XOR(mx, 0x041F));
            mx = fmaxf(mx, SWZ_XOR(mx, 0x081F));
            mx = fmaxf(mx, SWZ_XOR(mx, 0x101F));
            mx = fmaxf(mx, SWZ_XOR(mx, 0x201F));
            float mn    = fmaxf(m[r], mx);
            float alpha = __expf(m[r] - mn);
            float rsum  = 0.0f;
#pragma unroll
            for (int nt = 0; nt < 4; ++nt) {
                float p = __expf(s[nt][r] - mn);
                s[nt][r] = p;
                rsum += p;
            }
            rsum += SWZ_XOR(rsum, 0x041F);
            rsum += SWZ_XOR(rsum, 0x081F);
            rsum += SWZ_XOR(rsum, 0x101F);
            rsum += SWZ_XOR(rsum, 0x201F);
            l[r] = l[r] * alpha + rsum;
            m[r] = mn;
#pragma unroll
            for (int j = 0; j < 4; ++j) o[j][r] *= alpha;
        }

        // restage P as bf16 into this wave's private LDS region (A-layout)
        bf16* pw = &Ps[wid * 16 * LP];
#pragma unroll
        for (int nt = 0; nt < 4; ++nt)
#pragma unroll
            for (int r = 0; r < 8; ++r)
                pw[(rsel + r) * LP + nt * 16 + colL] = (bf16)s[nt][r];
        asm volatile("s_wait_dscnt 0" ::: "memory");    // in-wave LDS RAW

        // O += P @ V   (keys=64 -> 2 k-steps, hd -> 4 n-tiles)
#pragma unroll
        for (int j = 0; j < 4; ++j)
#pragma unroll
            for (int ks = 0; ks < 2; ++ks) {
                int ko2 = ks * 32 + koff;
                v16bf af = load_frag(&pw[colL * LP], ko2);
                v16bf bf = load_frag(&VTs[(j * 16 + colL) * LV], ko2);
                o[j] = WMMA_BF16(af, bf, o[j]);
            }
    }

    // normalize and emit ctx (bf16) in [B*S][1024] layout
#pragma unroll
    for (int j = 0; j < 4; ++j)
#pragma unroll
        for (int r = 0; r < 8; ++r) {
            int row = q0 + qr0 + rsel + r;
            ctx[(size_t)(b * S + row) * 1024 + h * HD + j * 16 + colL] =
                (bf16)(o[j][r] / l[r]);
        }
}

// ---------------------------------------------------------------------------
extern "C" void kernel_launch(void* const* d_in, const int* in_sizes, int n_in,
                              void* d_out, int out_size, void* d_ws, size_t ws_size,
                              hipStream_t stream) {
    const float* x    = (const float*)d_in[0];
    const float* Wqkv = (const float*)d_in[1];
    const float* bqkv = (const float*)d_in[2];
    const float* Wout = (const float*)d_in[3];
    const float* bout = (const float*)d_in[4];
    float* out = (float*)d_out;

    constexpr int M = 4096, D = 1024, N3 = 3072;
    char* ws = (char*)d_ws;
    // layout (bytes): xb/ctx alias (xb dead after GEMM1)
    bf16* xb   = (bf16*)(ws);                               //  8 MiB
    bf16* ctxb = xb;                                        //  8 MiB (alias)
    bf16* wqT  = (bf16*)(ws + (size_t)8  * 1024 * 1024);    //  6 MiB
    bf16* woT  = (bf16*)(ws + (size_t)14 * 1024 * 1024);    //  2 MiB
    bf16* qkvb = (bf16*)(ws + (size_t)16 * 1024 * 1024);    // 24 MiB  (total 40)

    f32_to_bf16_kernel<<<2048, 256, 0, stream>>>(x, xb, (size_t)M * D);
    f32_to_bf16_T_kernel<<<2048, 256, 0, stream>>>(Wqkv, wqT, D, N3);
    f32_to_bf16_T_kernel<<<2048, 256, 0, stream>>>(Wout, woT, D, D);

    gemm_bf16_kernel<true><<<dim3(M / 128, N3 / 128), 256, 0, stream>>>(
        xb, wqT, bqkv, qkvb, M, N3, D, N3);

    attn_kernel<<<512, 256, 0, stream>>>(qkvb, ctxb);

    gemm_bf16_kernel<false><<<dim3(M / 128, D / 128), 256, 0, stream>>>(
        ctxb, woT, bout, out, M, D, D, D);
}
